// RSSM_29368986370533
// MI455X (gfx1250) — matrix operand order repaired and checked by reading, not compiled, not run on hardware
//
#include <hip/hip_runtime.h>
#include <stdint.h>

typedef __bf16 bf16;
typedef __attribute__((ext_vector_type(16))) __bf16 v16bf;
typedef __attribute__((ext_vector_type(8)))  __bf16 v8bf;
typedef __attribute__((ext_vector_type(8)))  float  v8f;

#define B_      16
#define T_      64
#define STOCH_  32
#define DISC_   32
#define DETER_  2048
#define HIDDEN_ 1024
#define EMBED_  1024
#define OUTW    6144   // post_stoch(1024) deter(2048) post_logits(1024) prior_stoch(1024) prior_logits(1024)
#define LN_EPS  1e-3f

__device__ __forceinline__ bf16 f2bf(float f) {
  union { float f; uint32_t u; } x; x.f = f;
  uint32_t r = x.u + 0x7FFFu + ((x.u >> 16) & 1u);   // round-to-nearest-even
  union { uint16_t s; bf16 b; } y;
  y.s = (uint16_t)(r >> 16);
  return y.b;
}

// ---------------------------------------------------------------------------
// Repack fp32 weight W[Ksrc][N] (row-major, as in "x @ W") into bf16 WMMA
// B-operand tiles: layout [N/16][Kpad/32][32 lanes][16 elems].
// Per lane l (n = nt*16 + (l&15), h = l>>4), element e: k = kt*32 + 8*h + e (e<8),
// k = kt*32 + 16 + 8*h + (e-8) (e>=8) -- matches the A-fragment k order.
// ---------------------------------------------------------------------------
__global__ void repack_weight(const float* __restrict__ W, bf16* __restrict__ Wp,
                              int Ksrc, int N, int Kpad) {
  size_t idx = (size_t)blockIdx.x * blockDim.x + threadIdx.x;
  int KT = Kpad >> 5;
  size_t total = (size_t)(N >> 4) * KT * 512;
  if (idx >= total) return;
  int e    = (int)(idx & 15);
  int lane = (int)((idx >> 4) & 31);
  size_t tile = idx >> 9;
  int kt = (int)(tile % KT);
  int nt = (int)(tile / KT);
  int h = lane >> 4;
  int n = nt * 16 + (lane & 15);
  int k = kt * 32 + 8 * h + (e < 8 ? e : 8 + e);
  float v = (k < Ksrc) ? W[(size_t)k * N + n] : 0.0f;
  Wp[idx] = f2bf(v);
}

// ---------------------------------------------------------------------------
// WMMA GEMM: out[16][N] (fp32) = A[16][K] (bf16) * Wp (packed bf16), K%32==0.
// A is a concat of two sources split at K0 (multiple of 32): two-phase K loop
// so there are NO per-iteration source selects. Each wave owns TWO adjacent
// 16x16 output tiles, sharing the A fragment (2 WMMAs per k-step).
// A-frag per ISA: lane (m=l&15,h=l>>4) elems 0..7 = A[m][kt*32+8h .. +7],
// elems 8..15 = A[m][kt*32+16+8h .. +7] -> two contiguous 16B loads.
// ---------------------------------------------------------------------------
__global__ __launch_bounds__(32)
void gemm_wmma(const bf16* __restrict__ A0, int lda0, int K0,
               const bf16* __restrict__ A1, int lda1,
               const bf16* __restrict__ Wp,
               float* __restrict__ out, int K, int N) {
  const int np   = blockIdx.x;           // pair of tile columns
  const int lane = threadIdx.x;
  const int m = lane & 15;
  const int h = lane >> 4;
  const int KT = K >> 5;
  int KT0 = K0 >> 5;
  if (KT0 > KT) KT0 = KT;
  v8f c0 = {0.f,0.f,0.f,0.f,0.f,0.f,0.f,0.f};
  v8f c1 = {0.f,0.f,0.f,0.f,0.f,0.f,0.f,0.f};
  const bf16* wt0 = Wp + (size_t)(2 * np)     * KT * 512 + lane * 16;
  const bf16* wt1 = Wp + (size_t)(2 * np + 1) * KT * 512 + lane * 16;

  const bf16* pa = A0 + (size_t)m * lda0 + 8 * h;
  for (int i = 0; i < KT0; ++i) {
    v8bf alo = *(const v8bf*)(pa);
    v8bf ahi = *(const v8bf*)(pa + 16);
    v16bf a;
#pragma unroll
    for (int j = 0; j < 8; ++j) { a[j] = alo[j]; a[8 + j] = ahi[j]; }
    v16bf b0 = *(const v16bf*)wt0;
    v16bf b1 = *(const v16bf*)wt1;
    pa += 32; wt0 += 512; wt1 += 512;
    c0 = __builtin_amdgcn_wmma_f32_16x16x32_bf16(false, a, false, b0, (short)0, c0, false, false);
    c1 = __builtin_amdgcn_wmma_f32_16x16x32_bf16(false, a, false, b1, (short)0, c1, false, false);
  }
  pa = A1 + (size_t)m * lda1 + 8 * h;
  for (int i = KT0; i < KT; ++i) {
    v8bf alo = *(const v8bf*)(pa);
    v8bf ahi = *(const v8bf*)(pa + 16);
    v16bf a;
#pragma unroll
    for (int j = 0; j < 8; ++j) { a[j] = alo[j]; a[8 + j] = ahi[j]; }
    v16bf b0 = *(const v16bf*)wt0;
    v16bf b1 = *(const v16bf*)wt1;
    pa += 32; wt0 += 512; wt1 += 512;
    c0 = __builtin_amdgcn_wmma_f32_16x16x32_bf16(false, a, false, b0, (short)0, c0, false, false);
    c1 = __builtin_amdgcn_wmma_f32_16x16x32_bf16(false, a, false, b1, (short)0, c1, false, false);
  }
  // D layout: VGPR r -> row m=r+8h, col = tile_base + (lane&15)
  float* orow = out + (size_t)(8 * h) * N + np * 32 + m;
#pragma unroll
  for (int r = 0; r < 8; ++r) {
    orow[(size_t)r * N]      = c0[r];
    orow[(size_t)r * N + 16] = c1[r];
  }
}

// ---------------------------------------------------------------------------
// Fused LayerNorm (over N) + SiLU, emit bf16 activation row.
// One block (256 thr = 8 waves) per batch row.
// ---------------------------------------------------------------------------
__global__ void ln_silu(const float* __restrict__ raw,
                        const float* __restrict__ s, const float* __restrict__ bia,
                        bf16* __restrict__ outbf, int N) {
  int b = blockIdx.x;
  const float* row = raw + (size_t)b * N;
  float sum = 0.f, sq = 0.f;
  for (int i = threadIdx.x; i < N; i += blockDim.x) { float v = row[i]; sum += v; sq += v * v; }
  __shared__ float s1[8], s2[8];
  for (int off = 16; off > 0; off >>= 1) { sum += __shfl_down(sum, off, 32); sq += __shfl_down(sq, off, 32); }
  int wv = threadIdx.x >> 5, ln = threadIdx.x & 31;
  if (ln == 0) { s1[wv] = sum; s2[wv] = sq; }
  __syncthreads();
  if (wv == 0) {
    sum = (ln < 8) ? s1[ln] : 0.f; sq = (ln < 8) ? s2[ln] : 0.f;
    for (int off = 16; off > 0; off >>= 1) { sum += __shfl_down(sum, off, 32); sq += __shfl_down(sq, off, 32); }
    if (ln == 0) { s1[0] = sum; s2[0] = sq; }
  }
  __syncthreads();
  float mean = s1[0] / N;
  float var  = s2[0] / N - mean * mean;
  float inv  = rsqrtf(var + LN_EPS);
  for (int i = threadIdx.x; i < N; i += blockDim.x) {
    float y = (row[i] - mean) * inv * s[i] + bia[i];
    float sig = 1.f / (1.f + expf(-y));
    outbf[(size_t)b * N + i] = f2bf(y * sig);
  }
}

// ---------------------------------------------------------------------------
// GRU: LN over 6144, split reset/cand/update, update deter (fp32 carry +
// bf16 GEMM copy), write deter slice of d_out. One block per batch row.
// ---------------------------------------------------------------------------
__global__ void gru_update(const float* __restrict__ raw,
                           const float* __restrict__ gs, const float* __restrict__ gb,
                           float* __restrict__ deter_f, bf16* __restrict__ deter_b,
                           float* __restrict__ out, int t) {
  int b = blockIdx.x;
  const float* row = raw + (size_t)b * (3 * DETER_);
  float sum = 0.f, sq = 0.f;
  for (int i = threadIdx.x; i < 3 * DETER_; i += blockDim.x) { float v = row[i]; sum += v; sq += v * v; }
  __shared__ float s1[8], s2[8];
  for (int off = 16; off > 0; off >>= 1) { sum += __shfl_down(sum, off, 32); sq += __shfl_down(sq, off, 32); }
  int wv = threadIdx.x >> 5, ln = threadIdx.x & 31;
  if (ln == 0) { s1[wv] = sum; s2[wv] = sq; }
  __syncthreads();
  if (wv == 0) {
    sum = (ln < 8) ? s1[ln] : 0.f; sq = (ln < 8) ? s2[ln] : 0.f;
    for (int off = 16; off > 0; off >>= 1) { sum += __shfl_down(sum, off, 32); sq += __shfl_down(sq, off, 32); }
    if (ln == 0) { s1[0] = sum; s2[0] = sq; }
  }
  __syncthreads();
  float mean = s1[0] / (3.f * DETER_);
  float var  = s2[0] / (3.f * DETER_) - mean * mean;
  float inv  = rsqrtf(var + LN_EPS);
  float* od = out + ((size_t)(b * T_ + t)) * OUTW + 1024;  // deter slot
  for (int n = threadIdx.x; n < DETER_; n += blockDim.x) {
    float p0 = (row[n           ] - mean) * inv * gs[n           ] + gb[n           ];
    float p1 = (row[n + DETER_  ] - mean) * inv * gs[n + DETER_  ] + gb[n + DETER_  ];
    float p2 = (row[n + 2*DETER_] - mean) * inv * gs[n + 2*DETER_] + gb[n + 2*DETER_];
    float r = 1.f / (1.f + expf(-p0));
    float cnd = tanhf(r * p1);
    float u = 1.f / (1.f + expf(-(p2 - 1.f)));
    float d = u * cnd + (1.f - u) * deter_f[(size_t)b * DETER_ + n];
    deter_f[(size_t)b * DETER_ + n] = d;
    deter_b[(size_t)b * DETER_ + n] = f2bf(d);
    od[n] = d;
  }
}

// ---------------------------------------------------------------------------
// Softmax(32) + unimix + gumbel argmax one-hot. One wave per (group s, row b).
// Forward value of straight-through sample is exactly the one-hot.
// uptr==null -> mode (no gumbel). outp==null -> init pass (no output write).
// ---------------------------------------------------------------------------
__global__ void stats_sample(const float* __restrict__ raw, const float* __restrict__ bias,
                             const float* __restrict__ uptr, float* __restrict__ outp,
                             int off_logits, int off_onehot,
                             bf16* __restrict__ carry, int t) {
  int s = blockIdx.x, b = blockIdx.y, l = threadIdx.x;
  float v = raw[(size_t)b * (STOCH_ * DISC_) + s * DISC_ + l] + bias[s * DISC_ + l];
  float mx = v;
  for (int off = 16; off > 0; off >>= 1) mx = fmaxf(mx, __shfl_xor(mx, off, 32));
  float e = expf(v - mx);
  float sm = e;
  for (int off = 16; off > 0; off >>= 1) sm += __shfl_xor(sm, off, 32);
  float p  = 0.99f * (e / sm) + 0.01f / 32.f;
  float lg = logf(p);
  float g = 0.f;
  if (uptr) {
    float u = uptr[(((size_t)b * T_ + t) * STOCH_ + s) * DISC_ + l];
    g = -logf(-logf(u + 1e-8f) + 1e-8f);
  }
  float samp = lg + g;
  float bv = samp; int bi = l;
  for (int off = 16; off > 0; off >>= 1) {
    float ov = __shfl_xor(bv, off, 32);
    int   oi = __shfl_xor(bi, off, 32);
    if (ov > bv || (ov == bv && oi < bi)) { bv = ov; bi = oi; }
  }
  float oh = (l == bi) ? 1.f : 0.f;
  if (outp) {
    float* base = outp + ((size_t)(b * T_ + t)) * OUTW;
    base[off_logits + s * DISC_ + l] = lg;
    base[off_onehot + s * DISC_ + l] = oh;
  }
  if (carry) carry[(size_t)b * (STOCH_ * DISC_) + s * DISC_ + l] = f2bf(oh);
}

// ---------------------------------------------------------------------------
// Per-step reset + pack A_inp[16][1056] = [stoch(1024) | action*(1-f)(6) | 0(26)]
// and apply is_first reset to deter (f in {0,1} -> exact select).
// ---------------------------------------------------------------------------
__global__ void reset_pack(const bf16* __restrict__ stoch_carry, const bf16* __restrict__ stoch0,
                           const float* __restrict__ deter0, const float* __restrict__ action,
                           const int* __restrict__ is_first,
                           float* __restrict__ deter_f, bf16* __restrict__ deter_b,
                           bf16* __restrict__ A_inp, int t) {
  int b = blockIdx.y;
  int f = (t == 0) ? 1 : (is_first[b * T_ + t] != 0 ? 1 : 0);
  int i = blockIdx.x * blockDim.x + threadIdx.x;
  if (i < DETER_) {
    if (f) {
      float d = deter0[i];
      deter_f[(size_t)b * DETER_ + i] = d;
      deter_b[(size_t)b * DETER_ + i] = f2bf(d);
    }
    return;
  }
  int j = i - DETER_;
  if (j >= 1056) return;
  bf16 val;
  if (j < 1024) {
    val = f ? stoch0[(size_t)b * 1024 + j] : stoch_carry[(size_t)b * 1024 + j];
  } else if (j < 1030) {
    float a = f ? 0.f : action[((size_t)b * T_ + t) * 6 + (j - 1024)];
    val = f2bf(a);
  } else {
    val = f2bf(0.f);  // K padding
  }
  A_inp[(size_t)b * 1056 + j] = val;
}

__global__ void cvt_bf16(const float* __restrict__ in, bf16* __restrict__ out, size_t n) {
  size_t i = (size_t)blockIdx.x * blockDim.x + threadIdx.x;
  if (i < n) out[i] = f2bf(in[i]);
}

__global__ void init_deter(const float* __restrict__ W_init, float* __restrict__ deter0_f,
                           float* __restrict__ deter_f, bf16* __restrict__ deter_b) {
  int n = blockIdx.x * blockDim.x + threadIdx.x;
  if (n >= DETER_) return;
  float d = tanhf(W_init[n]);
  deter0_f[n] = d;
  bf16 db = f2bf(d);
  for (int b = 0; b < B_; ++b) {
    deter_f[(size_t)b * DETER_ + n] = d;
    deter_b[(size_t)b * DETER_ + n] = db;
  }
}

// ---------------------------------------------------------------------------
extern "C" void kernel_launch(void* const* d_in, const int* in_sizes, int n_in,
                              void* d_out, int out_size, void* d_ws, size_t ws_size,
                              hipStream_t stream) {
  const float* embed    = (const float*)d_in[0];
  const float* action   = (const float*)d_in[1];
  const int*   is_first = (const int*)  d_in[2];
  const float* u_prior  = (const float*)d_in[3];
  const float* u_post   = (const float*)d_in[4];
  const float* W_init   = (const float*)d_in[5];
  const float* inp_W    = (const float*)d_in[6];
  const float* inp_s    = (const float*)d_in[7];
  const float* inp_b    = (const float*)d_in[8];
  const float* gru_W    = (const float*)d_in[9];
  const float* gru_s    = (const float*)d_in[10];
  const float* gru_b    = (const float*)d_in[11];
  const float* imgout_W = (const float*)d_in[12];
  const float* imgout_s = (const float*)d_in[13];
  const float* imgout_b = (const float*)d_in[14];
  const float* obsout_W = (const float*)d_in[15];
  const float* obsout_s = (const float*)d_in[16];
  const float* obsout_b = (const float*)d_in[17];
  const float* imgst_W  = (const float*)d_in[18];
  const float* imgst_b  = (const float*)d_in[19];
  const float* obsst_W  = (const float*)d_in[20];
  const float* obsst_b  = (const float*)d_in[21];
  float* outp = (float*)d_out;

  // --- carve workspace (256B aligned slices) ---
  char* cur = (char*)d_ws;
  auto carve = [&](size_t bytes) -> void* {
    void* r = (void*)cur;
    cur += (bytes + 255) & ~(size_t)255;
    return r;
  };
  bf16* Wp_inp    = (bf16*)carve((size_t)(1056/32)*(1024/16)*512 * 2);
  bf16* Wp_gru    = (bf16*)carve((size_t)(3072/32)*(6144/16)*512 * 2);
  bf16* Wp_imgout = (bf16*)carve((size_t)(2048/32)*(1024/16)*512 * 2);
  bf16* Wp_obsout = (bf16*)carve((size_t)(3072/32)*(1024/16)*512 * 2);
  bf16* Wp_imgst  = (bf16*)carve((size_t)(1024/32)*(1024/16)*512 * 2);
  bf16* Wp_obsst  = (bf16*)carve((size_t)(1024/32)*(1024/16)*512 * 2);
  bf16* embed_bf  = (bf16*)carve((size_t)B_*T_*EMBED_ * 2);
  bf16* stoch_carry = (bf16*)carve((size_t)B_*1024 * 2);
  bf16* stoch0    = (bf16*)carve((size_t)B_*1024 * 2);
  bf16* A_inp     = (bf16*)carve((size_t)B_*1056 * 2);
  bf16* x_bf      = (bf16*)carve((size_t)B_*HIDDEN_ * 2);
  bf16* y_bf      = (bf16*)carve((size_t)B_*HIDDEN_ * 2);
  bf16* z_bf      = (bf16*)carve((size_t)B_*HIDDEN_ * 2);
  bf16* deter_b   = (bf16*)carve((size_t)B_*DETER_ * 2);
  float* deter_f  = (float*)carve((size_t)B_*DETER_ * 4);
  float* deter0_f = (float*)carve((size_t)DETER_ * 4);
  float* raw_s    = (float*)carve((size_t)B_*1024 * 4);
  float* raw_g    = (float*)carve((size_t)B_*3*DETER_ * 4);
  (void)ws_size; (void)in_sizes; (void)n_in; (void)out_size;

  // --- pack weights to bf16 WMMA tiles (per-launch, deterministic) ---
  auto rp = [&](const float* W, bf16* Wp, int Ksrc, int N, int Kpad) {
    size_t total = (size_t)(N / 16) * (Kpad / 32) * 512;
    repack_weight<<<dim3((unsigned)((total + 255) / 256)), 256, 0, stream>>>(W, Wp, Ksrc, N, Kpad);
  };
  rp(inp_W,    Wp_inp,    1030, 1024, 1056);
  rp(gru_W,    Wp_gru,    3072, 6144, 3072);
  rp(imgout_W, Wp_imgout, 2048, 1024, 2048);
  rp(obsout_W, Wp_obsout, 3072, 1024, 3072);
  rp(imgst_W,  Wp_imgst,  1024, 1024, 1024);
  rp(obsst_W,  Wp_obsst,  1024, 1024, 1024);
  {
    size_t n = (size_t)B_ * T_ * EMBED_;
    cvt_bf16<<<dim3((unsigned)((n + 255) / 256)), 256, 0, stream>>>(embed, embed_bf, n);
  }

  // --- learned initial state: deter0 = tanh(W_init); stoch0 = mode(prior(deter0))
  init_deter<<<8, 256, 0, stream>>>(W_init, deter0_f, deter_f, deter_b);
  gemm_wmma<<<HIDDEN_/32, 32, 0, stream>>>(deter_b, DETER_, DETER_, deter_b, DETER_,
                                           Wp_imgout, raw_s, DETER_, HIDDEN_);
  ln_silu<<<B_, 256, 0, stream>>>(raw_s, imgout_s, imgout_b, y_bf, HIDDEN_);
  gemm_wmma<<<1024/32, 32, 0, stream>>>(y_bf, HIDDEN_, HIDDEN_, y_bf, HIDDEN_,
                                        Wp_imgst, raw_s, HIDDEN_, 1024);
  stats_sample<<<dim3(STOCH_, B_), 32, 0, stream>>>(raw_s, imgst_b, nullptr, nullptr, 0, 0, stoch0, 0);

  // --- sequential scan over T (stream ordering resolves the recurrence) ---
  for (int t = 0; t < T_; ++t) {
    reset_pack<<<dim3(13, B_), 256, 0, stream>>>(stoch_carry, stoch0, deter0_f, action,
                                                 is_first, deter_f, deter_b, A_inp, t);
    // x = silu(LN([stoch|a] @ inp_W))
    gemm_wmma<<<HIDDEN_/32, 32, 0, stream>>>(A_inp, 1056, 1056, A_inp, 1056,
                                             Wp_inp, raw_s, 1056, HIDDEN_);
    ln_silu<<<B_, 256, 0, stream>>>(raw_s, inp_s, inp_b, x_bf, HIDDEN_);
    // GRU: parts = LN([x|deter] @ gru_W)
    gemm_wmma<<<(3*DETER_)/32, 32, 0, stream>>>(x_bf, HIDDEN_, HIDDEN_, deter_b, DETER_,
                                                Wp_gru, raw_g, 3072, 3*DETER_);
    gru_update<<<B_, 256, 0, stream>>>(raw_g, gru_s, gru_b, deter_f, deter_b, outp, t);
    // prior head
    gemm_wmma<<<HIDDEN_/32, 32, 0, stream>>>(deter_b, DETER_, DETER_, deter_b, DETER_,
                                             Wp_imgout, raw_s, DETER_, HIDDEN_);
    ln_silu<<<B_, 256, 0, stream>>>(raw_s, imgout_s, imgout_b, y_bf, HIDDEN_);
    gemm_wmma<<<1024/32, 32, 0, stream>>>(y_bf, HIDDEN_, HIDDEN_, y_bf, HIDDEN_,
                                          Wp_imgst, raw_s, HIDDEN_, 1024);
    stats_sample<<<dim3(STOCH_, B_), 32, 0, stream>>>(raw_s, imgst_b, u_prior, outp,
                                                      5120, 4096, nullptr, t);
    // posterior head: A = [deter | embed_t]
    gemm_wmma<<<HIDDEN_/32, 32, 0, stream>>>(deter_b, DETER_, DETER_,
                                             embed_bf + (size_t)t * EMBED_, T_ * EMBED_,
                                             Wp_obsout, raw_s, 3072, HIDDEN_);
    ln_silu<<<B_, 256, 0, stream>>>(raw_s, obsout_s, obsout_b, z_bf, HIDDEN_);
    gemm_wmma<<<1024/32, 32, 0, stream>>>(z_bf, HIDDEN_, HIDDEN_, z_bf, HIDDEN_,
                                          Wp_obsst, raw_s, HIDDEN_, 1024);
    stats_sample<<<dim3(STOCH_, B_), 32, 0, stream>>>(raw_s, obsst_b, u_post, outp,
                                                      3072, 0, stoch_carry, t);
  }
}
